// GroupedQueryAttention_77206332113760
// MI455X (gfx1250) — matrix operand order repaired and checked by reading, not compiled, not run on hardware
//
#include <hip/hip_runtime.h>

typedef _Float16 half_t;
typedef __attribute__((ext_vector_type(16))) _Float16 v16h;
typedef __attribute__((ext_vector_type(2)))  _Float16 v2h;
typedef __attribute__((ext_vector_type(8)))  float    v8f;
typedef unsigned int v4u __attribute__((ext_vector_type(4)));
typedef int          v4i __attribute__((ext_vector_type(4)));
typedef int          v8i __attribute__((ext_vector_type(8)));

#define HEADS   16
#define DIMS    64
#define IN_DIMS 1024
#define BATCH   2
#define SEQ     2048

// ---------------- CDNA5 feature detection ----------------
#if defined(__has_builtin)
#  if __has_builtin(__builtin_amdgcn_tensor_load_to_lds)
#    define HAVE_TDM 1
#  endif
#endif
#ifndef HAVE_TDM
#  define HAVE_TDM 0
#endif

#if HAVE_TDM
#  if __has_builtin(__builtin_amdgcn_s_wait_tensorcnt)
#    define WAIT_TENSORCNT(n) __builtin_amdgcn_s_wait_tensorcnt(n)
#  else
#    define WAIT_TENSORCNT(n) asm volatile("s_wait_tensorcnt %0" ::"i"(n) : "memory")
#  endif
#endif

union Frag32B { uint4 q[2]; v16h h; };
union H8 { half_t h[8]; uint4 u; };

static __device__ __forceinline__ v8f wmma_f16(v16h a, v16h b, v8f c) {
  return __builtin_amdgcn_wmma_f32_16x16x32_f16(false, a, false, b, (short)0, c,
                                                false, false);
}

// A fragment: 16(M) x 32(K) row-major, leading dim ld halves. 2 x ds/global b128.
static __device__ __forceinline__ v16h load_frag_a(const half_t* p, int ld) {
  const int lane = threadIdx.x & 31;
  const int m    = lane & 15;
  const int khi  = (lane & 16) ? 8 : 0;
  const half_t* row = p + m * ld + khi;
  Frag32B f;
  f.q[0] = *(const uint4*)(row);
  f.q[1] = *(const uint4*)(row + 16);
  return f.h;
}

// B fragment from K-contiguous storage: B[k][n] at p[n*32 + k]. 2 x b128.
static __device__ __forceinline__ v16h load_frag_b_kcontig(const half_t* p) {
  const int lane = threadIdx.x & 31;
  const int n    = lane & 15;
  const int kb   = (lane & 16) ? 8 : 0;
  const half_t* src = p + n * 32 + kb;
  Frag32B f;
  f.q[0] = *(const uint4*)(src);
  f.q[1] = *(const uint4*)(src + 16);
  return f.h;
}

#if HAVE_TDM
// ---- Tensor Data Mover: 2-D tile load, rows of 64B (16 x 4B units), 128 rows,
// row stride = 1024 halves (2048B = 512 units), into contiguous LDS.
static __device__ __forceinline__ void tdm_load_tile(const half_t* src,
                                                     unsigned lds_byte,
                                                     int tensor_rows) {
  unsigned long long ga = (unsigned long long)(const void*)src;
  const unsigned td0 = 512u;   // tensor_dim0 in 4B units (full K row)
  const unsigned tile0 = 16u;  // 64B per row
  const unsigned td1 = (unsigned)tensor_rows;
  v4u g0;
  g0[0] = 1u;                                   // count=1, user mode
  g0[1] = lds_byte;                             // lds_addr
  g0[2] = (unsigned)(ga & 0xffffffffu);         // global_addr[31:0]
  g0[3] = (unsigned)((ga >> 32) & 0x01ffffffu) | (2u << 30);  // addr[56:32]|type=2
  v8i g1;
  g1[0] = (int)(2u << 16);                      // data_size = 4B
  g1[1] = (int)((td0 & 0xffffu) << 16);         // tensor_dim0 lo
  g1[2] = (int)((td0 >> 16) | ((td1 & 0xffffu) << 16));
  g1[3] = (int)((td1 >> 16) | (tile0 << 16));   // tile_dim0
  g1[4] = 128;                                  // tile_dim1 = 128 rows
  g1[5] = 512;                                  // tensor_dim0_stride lo32 (units)
  g1[6] = 0;
  g1[7] = 0;
  v4i z4 = {0, 0, 0, 0};
#if defined(__clang_major__) && __clang_major__ >= 23
  v8i z8 = {0, 0, 0, 0, 0, 0, 0, 0};
  __builtin_amdgcn_tensor_load_to_lds(g0, g1, z4, z4, z8, 0);
#else
  __builtin_amdgcn_tensor_load_to_lds(g0, g1, z4, z4, 0);
#endif
}
#endif

// Fallback cooperative tile copy (256 threads): 128 rows x 32 halves
static __device__ __forceinline__ void copy_tile(half_t* dst, const half_t* src) {
  const int t = threadIdx.x;
  const int row = t >> 1, c = (t & 1) * 16;
  const uint4* s = (const uint4*)(src + (size_t)row * IN_DIMS + c);
  *(uint4*)&dst[row * 32 + c]     = s[0];
  *(uint4*)&dst[row * 32 + c + 8] = s[1];
}

// ------------------------------------------------------------------
// Prep: f32 -> f16 elementwise
// ------------------------------------------------------------------
__global__ void __launch_bounds__(256)
cvt_f16_kernel(const float* __restrict__ src, half_t* __restrict__ dst) {
  const int i = (blockIdx.x * 256 + threadIdx.x) * 8;
  float4 a = *(const float4*)(src + i);
  float4 b = *(const float4*)(src + i + 4);
  H8 o;
  o.h[0] = (half_t)a.x; o.h[1] = (half_t)a.y; o.h[2] = (half_t)a.z; o.h[3] = (half_t)a.w;
  o.h[4] = (half_t)b.x; o.h[5] = (half_t)b.y; o.h[6] = (half_t)b.z; o.h[7] = (half_t)b.w;
  *(uint4*)(dst + i) = o.u;
}

// ------------------------------------------------------------------
// Prep: transpose W[K=1024][N] f32 -> Wt[N][1024] f16 (K-contiguous panels)
// ------------------------------------------------------------------
__global__ void __launch_bounds__(256)
transpose_f16_kernel(const float* __restrict__ W, int N, half_t* __restrict__ Wt) {
  __shared__ half_t T[64][72];
  const int t = threadIdx.x;
  const int k0 = blockIdx.y * 64, n0 = blockIdx.x * 64;
  const int rr = t >> 4, cc = (t & 15) * 4;
#pragma unroll
  for (int i = 0; i < 4; ++i) {
    int r = rr + i * 16;
    float4 v = *(const float4*)(W + (size_t)(k0 + r) * N + n0 + cc);
    T[cc + 0][r] = (half_t)v.x;
    T[cc + 1][r] = (half_t)v.y;
    T[cc + 2][r] = (half_t)v.z;
    T[cc + 3][r] = (half_t)v.w;
  }
  __syncthreads();
#pragma unroll
  for (int i = 0; i < 4; ++i) {
    int rn = rr + i * 16;
    half_t* d = Wt + (size_t)(n0 + rn) * 1024 + k0 + cc;
    d[0] = T[rn][cc + 0];
    d[1] = T[rn][cc + 1];
    d[2] = T[rn][cc + 2];
    d[3] = T[rn][cc + 3];
  }
}

// ------------------------------------------------------------------
// GEMM (TN): C[M][ldc] = A[M][1024] (f16, K-contig) x Bt[N][1024]^T (f16)
// 128x128 block tile, K-step 32, 8 waves (2x4), TDM double-buffered staging
// ------------------------------------------------------------------
__global__ void __launch_bounds__(256)
gemm_tn_kernel(const half_t* __restrict__ A, const half_t* __restrict__ Bt,
               float* __restrict__ C, int ldc, int arows, int brows) {
  __shared__ __align__(16) half_t As[2][128 * 32];
  __shared__ __align__(16) half_t Bs[2][128 * 32];
  const int tid = threadIdx.x, wid = tid >> 5;
  const int gm0 = blockIdx.y * 128, gn0 = blockIdx.x * 128;
  const int waveM = wid >> 2, waveN = wid & 3;
  const half_t* Abase = A + (size_t)gm0 * IN_DIMS;
  const half_t* Bbase = Bt + (size_t)gn0 * IN_DIMS;
  v8f acc[4][2] = {};

#if HAVE_TDM
  const unsigned ldsA[2] = {(unsigned)(size_t)(void*)As[0], (unsigned)(size_t)(void*)As[1]};
  const unsigned ldsB[2] = {(unsigned)(size_t)(void*)Bs[0], (unsigned)(size_t)(void*)Bs[1]};
  if (wid == 0) {
    tdm_load_tile(Abase, ldsA[0], arows - gm0);
    tdm_load_tile(Bbase, ldsB[0], brows - gn0);
  }
  for (int i = 0; i < IN_DIMS / 32; ++i) {
    const int cur = i & 1;
    if (wid == 0) {
      if (i + 1 < IN_DIMS / 32) {
        tdm_load_tile(Abase + (i + 1) * 32, ldsA[cur ^ 1], arows - gm0);
        tdm_load_tile(Bbase + (i + 1) * 32, ldsB[cur ^ 1], brows - gn0);
        WAIT_TENSORCNT(2);  // retire stage i (TDM in-order per wave)
      } else {
        WAIT_TENSORCNT(0);
      }
    }
    __syncthreads();
    v16h b0 = load_frag_b_kcontig(&Bs[cur][(waveN * 32 + 0) * 32]);
    v16h b1 = load_frag_b_kcontig(&Bs[cur][(waveN * 32 + 16) * 32]);
#pragma unroll
    for (int fm = 0; fm < 4; ++fm) {
      v16h a = load_frag_a(&As[cur][(waveM * 64 + fm * 16) * 32], 32);
      acc[fm][0] = wmma_f16(a, b0, acc[fm][0]);
      acc[fm][1] = wmma_f16(a, b1, acc[fm][1]);
    }
    __syncthreads();
  }
#else
  for (int i = 0; i < IN_DIMS / 32; ++i) {
    copy_tile(As[0], Abase + i * 32);
    copy_tile(Bs[0], Bbase + i * 32);
    __syncthreads();
    v16h b0 = load_frag_b_kcontig(&Bs[0][(waveN * 32 + 0) * 32]);
    v16h b1 = load_frag_b_kcontig(&Bs[0][(waveN * 32 + 16) * 32]);
#pragma unroll
    for (int fm = 0; fm < 4; ++fm) {
      v16h a = load_frag_a(&As[0][(waveM * 64 + fm * 16) * 32], 32);
      acc[fm][0] = wmma_f16(a, b0, acc[fm][0]);
      acc[fm][1] = wmma_f16(a, b1, acc[fm][1]);
    }
    __syncthreads();
  }
#endif

  const int lane = tid & 31, n = lane & 15, hi = (lane & 16) ? 8 : 0;
#pragma unroll
  for (int fm = 0; fm < 4; ++fm)
#pragma unroll
    for (int fn = 0; fn < 2; ++fn)
#pragma unroll
      for (int r = 0; r < 8; ++r) {
        int row = gm0 + waveM * 64 + fm * 16 + r + hi;
        int col = gn0 + waveN * 32 + fn * 16 + n;
        C[(size_t)row * ldc + col] = acc[fm][fn][r];
      }
}

// ------------------------------------------------------------------
// RoPE + GQA pair-fold pack -> Qs/Kp/Vp f16 [B,16,S,32]
// ------------------------------------------------------------------
__global__ void __launch_bounds__(256)
rope_pack_kernel(const float* __restrict__ P, half_t* __restrict__ Qs,
                 half_t* __restrict__ Kp, half_t* __restrict__ Vp) {
  const int s = blockIdx.x & (SEQ - 1);
  const int b = blockIdx.x >> 11;
  const float* p = P + (size_t)blockIdx.x * 2048;
  const float pos = (float)(s + 1);

  for (int t = threadIdx.x; t < 512; t += 256) {
    {  // Q: rope pair i over dim 1024, fold (re+im), fold scale 0.5
      const int i = t;
      float sn, cs;
      float ang = pos * __powf(10000.f, -(float)i * (1.f / 512.f));
      __sincosf(ang, &sn, &cs);
      float xe = p[2 * i], xo = p[2 * i + 1];
      float re = xe * cs - xo * sn;
      float im = xe * sn + xo * cs;
      int h = i >> 5, j = i & 31;
      Qs[(((size_t)b * HEADS + h) * SEQ + s) * 32 + j] = (half_t)(0.5f * (re + im));
    }
    {  // K: rope over dim 512 ; V: copy
      const int m = t, ik = m >> 1;
      float sn, cs;
      float ang = pos * __powf(10000.f, -(float)ik * (1.f / 256.f));
      __sincosf(ang, &sn, &cs);
      float ke = p[1024 + 2 * ik], ko = p[1024 + 2 * ik + 1];
      float kv = (m & 1) ? (ke * sn + ko * cs) : (ke * cs - ko * sn);
      int h = m >> 5, j = m & 31;
      Kp[(((size_t)b * HEADS + h) * SEQ + s) * 32 + j] = (half_t)kv;
      Vp[(((size_t)b * HEADS + h) * SEQ + s) * 32 + j] = (half_t)p[1536 + m];
    }
  }
}

// ------------------------------------------------------------------
// Flash attention in the 32-dim folded head space
// ------------------------------------------------------------------
__global__ void __launch_bounds__(256)
attn_kernel(const half_t* __restrict__ Qs, const half_t* __restrict__ Kp,
            const half_t* __restrict__ Vp, half_t* __restrict__ attn) {
  __shared__ __align__(16) half_t Pl[8 * 16 * 32];
  const int tid = threadIdx.x, wid = tid >> 5, lane = tid & 31;
  const int qt = blockIdx.x & 15;
  const int bh = blockIdx.x >> 4;
  const int b = bh >> 4, h = bh & 15;
  const size_t base = (size_t)bh * SEQ * 32;
  const int q0 = qt * 128 + wid * 16;

  const int n  = lane & 15;
  const int hi = (lane & 16) ? 1 : 0;
  const int kb = hi * 16;

  const v16h qf = load_frag_a(Qs + base + (size_t)q0 * 32, 32);

  v8f acc0 = {}, acc1 = {};
  float mprev[8], lpart[8];
#pragma unroll
  for (int r = 0; r < 8; ++r) { mprev[r] = -3.0e38f; lpart[r] = 0.f; }
  half_t* myP = &Pl[wid * 512];

  for (int kv0 = 0; kv0 < SEQ; kv0 += 32) {
    if (kv0 + 32 < SEQ) {  // warm next tiles (global_prefetch_b8)
      __builtin_prefetch((const void*)(Kp + base + (size_t)(kv0 + 32) * 32), 0, 1);
      __builtin_prefetch((const void*)(Vp + base + (size_t)(kv0 + 32) * 32), 0, 1);
    }
    v16h kf0 = load_frag_b_kcontig(Kp + base + (size_t)kv0 * 32);
    v16h kf1 = load_frag_b_kcontig(Kp + base + (size_t)(kv0 + 16) * 32);
    v8f z = {};
    v8f s0 = wmma_f16(qf, kf0, z);
    v8f s1 = wmma_f16(qf, kf1, z);

    float alpha[8];
#pragma unroll
    for (int r = 0; r < 8; ++r) {
      float lm = fmaxf(s0[r], s1[r]);
      lm = fmaxf(lm, __shfl_xor(lm, 1));
      lm = fmaxf(lm, __shfl_xor(lm, 2));
      lm = fmaxf(lm, __shfl_xor(lm, 4));
      lm = fmaxf(lm, __shfl_xor(lm, 8));
      float mnew = fmaxf(mprev[r], lm);
      alpha[r] = __expf(mprev[r] - mnew);
      float p0 = __expf(s0[r] - mnew);
      float p1 = __expf(s1[r] - mnew);
      lpart[r] = lpart[r] * alpha[r] + p0 + p1;
      mprev[r] = mnew;
      int m = r + hi * 8;
      myP[m * 32 + n]      = (half_t)p0;
      myP[m * 32 + 16 + n] = (half_t)p1;
      acc0[r] *= alpha[r];
      acc1[r] *= alpha[r];
    }
    __syncthreads();
    v16h pf = load_frag_a(myP, 32);
    v16h vf0, vf1;
#pragma unroll
    for (int e = 0; e < 16; ++e) {
      vf0[e] = Vp[base + (size_t)(kv0 + hi * 8 + e + (e & 8)) * 32 + n];
      vf1[e] = Vp[base + (size_t)(kv0 + hi * 8 + e + (e & 8)) * 32 + 16 + n];
    }
    acc0 = wmma_f16(pf, vf0, acc0);
    acc1 = wmma_f16(pf, vf1, acc1);
    __syncthreads();
  }

#pragma unroll
  for (int r = 0; r < 8; ++r) {
    float l = lpart[r];
    l += __shfl_xor(l, 1);
    l += __shfl_xor(l, 2);
    l += __shfl_xor(l, 4);
    l += __shfl_xor(l, 8);
    float inv = 1.f / l;
    float o0 = acc0[r] * inv, o1 = acc1[r] * inv;
    int m = r + hi * 8;
    int srow = q0 + m;
    size_t ob = ((size_t)b * SEQ + srow) * 1024 + h * 64;
    v2h d0 = {(half_t)o0, (half_t)o0};  // pair-duplicated output dims
    v2h d1 = {(half_t)o1, (half_t)o1};
    *(v2h*)&attn[ob + 2 * n]      = d0;
    *(v2h*)&attn[ob + 32 + 2 * n] = d1;
  }
}

// ------------------------------------------------------------------
extern "C" void kernel_launch(void* const* d_in, const int* in_sizes, int n_in,
                              void* d_out, int out_size, void* d_ws, size_t ws_size,
                              hipStream_t stream) {
  const float* q  = (const float*)d_in[0];
  const float* Wq = (const float*)d_in[1];
  const float* Wk = (const float*)d_in[2];
  const float* Wv = (const float*)d_in[3];
  const float* Wo = (const float*)d_in[4];
  float* out = (float*)d_out;

  char* ws = (char*)d_ws;
  size_t off = 0;
  float* P = (float*)(ws + off);
  off += (size_t)BATCH * SEQ * 2048 * sizeof(float);             // 33.5 MB
  const size_t PACK = (size_t)BATCH * HEADS * SEQ * 32 * sizeof(half_t);
  half_t* Qs = (half_t*)(ws + off); off += PACK;                 // 4 MB
  half_t* Kp = (half_t*)(ws + off); off += PACK;                 // 4 MB
  half_t* Vp = (half_t*)(ws + off); off += PACK;                 // 4 MB
  half_t* attn = (half_t*)(ws + off);
  off += (size_t)BATCH * SEQ * 1024 * sizeof(half_t);            // 8 MB
  half_t* qh = (half_t*)(ws + off);
  off += (size_t)BATCH * SEQ * 1024 * sizeof(half_t);            // 8 MB
  half_t* Wt = (half_t*)(ws + off);
  off += (size_t)2048 * 1024 * sizeof(half_t);                   // 4 MB
  half_t* Wot = (half_t*)(ws + off);
  off += (size_t)1024 * 1024 * sizeof(half_t);                   // 2 MB

  // 1. activations to f16
  cvt_f16_kernel<<<(BATCH * SEQ * IN_DIMS) / (256 * 8), 256, 0, stream>>>(q, qh);
  // 2. weights to f16, transposed K-contiguous panels: Wt = [Wq|Wk|Wv]^T, Wot = Wo^T
  transpose_f16_kernel<<<dim3(16, 16), 256, 0, stream>>>(Wq, 1024, Wt);
  transpose_f16_kernel<<<dim3(8, 16), 256, 0, stream>>>(Wk, 512, Wt + (size_t)1024 * 1024);
  transpose_f16_kernel<<<dim3(8, 16), 256, 0, stream>>>(Wv, 512, Wt + (size_t)1536 * 1024);
  transpose_f16_kernel<<<dim3(16, 16), 256, 0, stream>>>(Wo, 1024, Wot);
  // 3. QKV projection
  gemm_tn_kernel<<<dim3(16, 32), 256, 0, stream>>>(qh, Wt, P, 2048, BATCH * SEQ, 2048);
  // 4. RoPE + pack
  rope_pack_kernel<<<BATCH * SEQ, 256, 0, stream>>>(P, Qs, Kp, Vp);
  // 5. attention
  attn_kernel<<<BATCH * HEADS * (SEQ / 128), 256, 0, stream>>>(Qs, Kp, Vp, attn);
  // 6. output projection
  gemm_tn_kernel<<<dim3(8, 32), 256, 0, stream>>>(attn, Wot, out, 1024, BATCH * SEQ, 1024);
}
